// Mamba2MIL_87582973100579
// MI455X (gfx1250) — matrix operand order repaired and checked by reading, not compiled, not run on hardware
//
#include <hip/hip_runtime.h>
#include <hip/hip_bf16.h>
#include <math.h>

// ---------------- model constants ----------------
#define LSEQ   10000
#define DMODEL 512
#define DINNER 1024
#define DSTATE 64
#define NHEADS 16
#define HEADDIM 64
#define DIP    2192      // 2*1024 + 2*64 + 16
#define DXBC   1152      // 1024 + 128
#define NB     3         // branches
#define LTOT   30000     // NB * LSEQ
#define QCH    64        // SSD chunk length
#define NCH    157       // ceil(LSEQ/QCH)

typedef __attribute__((ext_vector_type(16))) __bf16 bf16x16;
typedef __attribute__((ext_vector_type(8)))  __bf16 bf16x8;
typedef __attribute__((ext_vector_type(8)))  float  f32x8;

// ---------------- f32 -> bf16 convert ----------------
__global__ void f2bf_kernel(const float* __restrict__ src, __bf16* __restrict__ dst, long long n) {
    long long i = (long long)blockIdx.x * blockDim.x + threadIdx.x;
    long long stride = (long long)gridDim.x * blockDim.x;
    for (; i < n; i += stride) dst[i] = (__bf16)src[i];
}

// ---------------- generic bf16 WMMA GEMM: C[M,N] = act(A[M,K] @ W[N,K]^T + bias + resid) ----------------
__global__ __launch_bounds__(256) void gemm_bf16_wmma(
    const __bf16* __restrict__ A, const __bf16* __restrict__ W,
    const float* __restrict__ bias, const float* __restrict__ resid,
    float* __restrict__ Cf, __bf16* __restrict__ Cb,
    int M, int N, int K, int act, int tilesPerBatch, long long wBatchStride)
{
    const int lane = threadIdx.x & 31;
    const int wave = threadIdx.x >> 5;
    const int ntn  = N >> 4;
    const int ntiles = (M >> 4) * ntn;
    const int tile = blockIdx.x * 8 + wave;
    if (tile >= ntiles) return;
    const int tn = tile % ntn;
    const int tm = tile / ntn;

    const __bf16* Wp = W;
    if (tilesPerBatch > 0) Wp += (long long)(tm / tilesPerBatch) * wBatchStride;

    const int r    = lane & 15;
    const int half = lane >> 4;
    const long long arow = (long long)(tm * 16 + r) * K;
    const long long brow = (long long)(tn * 16 + r) * K;

    f32x8 acc = {0.f, 0.f, 0.f, 0.f, 0.f, 0.f, 0.f, 0.f};
    for (int k0 = 0; k0 < K; k0 += 32) {
        bf16x16 af = *(const bf16x16*)(A + arow + k0 + half * 16);
        bf16x16 bf = *(const bf16x16*)(Wp + brow + k0 + half * 16);
        acc = __builtin_amdgcn_wmma_f32_16x16x32_bf16(false, af, false, bf, (short)0, acc, false, false);
    }

    const int n = tn * 16 + r;
    const float bv = bias ? bias[n] : 0.f;
#pragma unroll
    for (int i = 0; i < 8; ++i) {
        const int m = tm * 16 + half * 8 + i;
        const long long o = (long long)m * N + n;
        float v = acc[i] + bv;
        if (resid) v += resid[o];
        if (act == 1) v = v > 0.f ? v : 0.f;
        else if (act == 2) v = tanhf(v);
        if (Cb) Cb[o] = (__bf16)v; else Cf[o] = v;
    }
}

// ---------------- LDS 16x16 WMMA tile over K=64: D += A[16rows]x64 . W[16rows]x64^T ----------------
__device__ inline f32x8 wmma_tile64(const __bf16* Ab, const __bf16* Wb, int tm, int tn, int lane, f32x8 acc) {
    const int r = lane & 15, half = lane >> 4;
    const __bf16* arow = Ab + (tm * 16 + r) * 64;
    const __bf16* wrow = Wb + (tn * 16 + r) * 64;
#pragma unroll
    for (int k0 = 0; k0 < 64; k0 += 32) {
        bf16x16 af = *(const bf16x16*)(arow + k0 + half * 16);
        bf16x16 wf = *(const bf16x16*)(wrow + k0 + half * 16);
        acc = __builtin_amdgcn_wmma_f32_16x16x32_bf16(false, af, false, wf, (short)0, acc, false, false);
    }
    return acc;
}

// ---------------- build flipped / token-transposed branch inputs ----------------
__global__ void prep_branches_kernel(float* __restrict__ H3) {
    long long i = (long long)blockIdx.x * blockDim.x + threadIdx.x;
    long long stride = (long long)gridDim.x * blockDim.x;
    const long long n = (long long)LSEQ * DMODEL;
    for (; i < n; i += stride) {
        const int t = (int)(i / DMODEL), d = (int)(i % DMODEL);
        const float v = H3[(long long)t * DMODEL + d];
        H3[((long long)LSEQ + t) * DMODEL + (DMODEL - 1 - d)] = v;
        const int c = t / 100, w = t % 100;
        H3[((long long)2 * LSEQ + (long long)w * 100 + c) * DMODEL + d] = v;
    }
}

// ---------------- causal depthwise conv (k=4) + bias + SiLU ----------------
__global__ void conv_silu_kernel(const __bf16* __restrict__ ZX, const float* __restrict__ conv_w,
                                 const float* __restrict__ conv_b, float* __restrict__ XC, int s)
{
    long long i = (long long)blockIdx.x * blockDim.x + threadIdx.x;
    long long stride = (long long)gridDim.x * blockDim.x;
    const long long n = (long long)NB * LSEQ * DXBC;
    for (; i < n; i += stride) {
        const int b = (int)(i / ((long long)LSEQ * DXBC));
        const long long rr = i % ((long long)LSEQ * DXBC);
        const int t = (int)(rr / DXBC), c = (int)(rr % DXBC);
        const int layer = 2 * b + s;
        float acc = conv_b[layer * DXBC + c];
#pragma unroll
        for (int k = 0; k < 4; ++k) {
            const int tt = t - 3 + k;
            if (tt >= 0)
                acc += (float)ZX[((long long)b * LSEQ + tt) * DIP + DINNER + c] *
                       conv_w[((long long)layer * DXBC + c) * 4 + k];
        }
        XC[((long long)b * LSEQ + t) * DXBC + c] = acc / (1.f + __expf(-acc));
    }
}

// ---------------- chunked SSD scan, all-WMMA ----------------
// grid: NB*NHEADS blocks, 256 threads (8 waves). Per chunk (Q=64):
//   M = causal-mask(G) o (C.B^T) ; Y = M.X + diag(exp(cum)).(C.S^T) + Dh*x ;
//   S = exp(cum_Q).S + Btilde^T.X  (state held transposed [p][n]; f32 master in regs)
__global__ __launch_bounds__(256) void ssd_chunk_kernel(
    const float* __restrict__ XC, const __bf16* __restrict__ ZX,
    const float* __restrict__ dt_bias, const float* __restrict__ A_log,
    const float* __restrict__ Dh, __bf16* __restrict__ Y, int s)
{
    const int b  = blockIdx.x / NHEADS;
    const int hh = blockIdx.x % NHEADS;
    const int layer = 2 * b + s;
    const int tid  = threadIdx.x;
    const int wave = tid >> 5;
    const int lane = tid & 31;
    const int r    = lane & 15;
    const int half = lane >> 4;

    __shared__ __align__(16) __bf16 sXT[64 * 64];   // X transposed: [p][t]
    __shared__ __align__(16) __bf16 sB [64 * 64];   // [t][n]
    __shared__ __align__(16) __bf16 sC [64 * 64];   // [t][n]
    __shared__ __align__(16) __bf16 sM [64 * 64];   // masked CB: [t][i]
    __shared__ __align__(16) __bf16 sBt[64 * 64];   // Btilde transposed: [n][t]
    __shared__ __align__(16) __bf16 sS [64 * 64];   // bf16 state copy: [p][n]
    __shared__ float sdt[64];
    __shared__ float sCum[64];

    const float Apar = -__expf(A_log[layer * NHEADS + hh]);
    const float dtb  = dt_bias[layer * NHEADS + hh];
    const float dh   = Dh[layer * NHEADS + hh];
    const long long xcb = (long long)b * LSEQ * DXBC;
    const long long zxb = (long long)b * LSEQ * DIP;
    const long long yb  = (long long)b * LSEQ * DINNER;

    // f32 master state: each wave owns tiles (wave*2, wave*2+1) of the 4x4 (p,n) grid
    float stReg[2][8];
#pragma unroll
    for (int u = 0; u < 2; ++u)
#pragma unroll
        for (int q = 0; q < 8; ++q) stReg[u][q] = 0.f;
    for (int e = tid; e < 4096; e += 256) sS[e] = (__bf16)0.f;
    __syncthreads();

    for (int ch = 0; ch < NCH; ++ch) {
        // ---- phase A: stage chunk into LDS (zero-pad past LSEQ) ----
        for (int e = tid; e < 4096; e += 256) {
            const int t = e >> 6, j = e & 63;
            const int tg = ch * QCH + t;
            float xv = 0.f, bv = 0.f, cv = 0.f;
            if (tg < LSEQ) {
                const long long xrow = xcb + (long long)tg * DXBC;
                xv = XC[xrow + hh * HEADDIM + j];
                bv = XC[xrow + DINNER + j];
                cv = XC[xrow + DINNER + DSTATE + j];
            }
            sXT[j * 64 + t] = (__bf16)xv;
            sB [t * 64 + j] = (__bf16)bv;
            sC [t * 64 + j] = (__bf16)cv;
        }
        if (tid < 64) {
            const int tg = ch * QCH + tid;
            float dtv = 0.f;
            if (tg < LSEQ) {
                const float dtr = (float)ZX[zxb + (long long)tg * DIP + DINNER + DXBC + hh] + dtb;
                dtv = (dtr > 20.f) ? dtr : log1pf(__expf(dtr));   // softplus
            }
            sdt[tid] = dtv;
        }
        __syncthreads();
        if (tid == 0) {                  // cumulative log-decay (64 adds, negligible)
            float c = 0.f;
            for (int t = 0; t < 64; ++t) { c += sdt[t] * Apar; sCum[t] = c; }
        }
        __syncthreads();
        // Btilde^T[n][t] = exp(cum_63 - cum_t)*dt_t*B[t][n]
        {
            const float cl = sCum[63];
            for (int e = tid; e < 4096; e += 256) {
                const int t = e >> 6, n = e & 63;
                const float sc = __expf(cl - sCum[t]) * sdt[t];
                sBt[n * 64 + t] = (__bf16)(sc * (float)sB[t * 64 + n]);
            }
        }
        __syncthreads();

        // ---- phase B: M = mask(G) o (C . B^T) ----
#pragma unroll
        for (int u = 0; u < 2; ++u) {
            const int tt = wave * 2 + u;
            const int tm = tt >> 2, tn = tt & 3;
            f32x8 acc = {0.f, 0.f, 0.f, 0.f, 0.f, 0.f, 0.f, 0.f};
            acc = wmma_tile64(sC, sB, tm, tn, lane, acc);
            const int ic = tn * 16 + r;
#pragma unroll
            for (int q = 0; q < 8; ++q) {
                const int tr = tm * 16 + half * 8 + q;
                const float g = (ic <= tr) ? __expf(sCum[tr] - sCum[ic]) * sdt[ic] : 0.f;
                sM[tr * 64 + ic] = (__bf16)(acc[q] * g);
            }
        }
        __syncthreads();

        // ---- phase C: Y = M.X + exp(cum_t)*(C.S^T) + dh*x ----
#pragma unroll
        for (int u = 0; u < 2; ++u) {
            const int tt = wave * 2 + u;
            const int tm = tt >> 2, tn = tt & 3;
            f32x8 ai = {0.f, 0.f, 0.f, 0.f, 0.f, 0.f, 0.f, 0.f};
            f32x8 as = {0.f, 0.f, 0.f, 0.f, 0.f, 0.f, 0.f, 0.f};
            ai = wmma_tile64(sM, sXT, tm, tn, lane, ai);
            as = wmma_tile64(sC, sS,  tm, tn, lane, as);
            const int p = tn * 16 + r;
#pragma unroll
            for (int q = 0; q < 8; ++q) {
                const int tr = tm * 16 + half * 8 + q;
                const int tg = ch * QCH + tr;
                if (tg < LSEQ) {
                    const float xv = (float)sXT[p * 64 + tr];
                    const float y = ai[q] + __expf(sCum[tr]) * as[q] + dh * xv;
                    Y[yb + (long long)tg * DINNER + hh * HEADDIM + p] = (__bf16)y;
                }
            }
        }
        __syncthreads();

        // ---- phase D: S = exp(cum_63)*S + Btilde^T . X   (output [p][n]) ----
        {
            const float atot = __expf(sCum[63]);
#pragma unroll
            for (int u = 0; u < 2; ++u) {
                const int tt = wave * 2 + u;
                const int tm = tt >> 2, tn = tt & 3;
                f32x8 acc = {0.f, 0.f, 0.f, 0.f, 0.f, 0.f, 0.f, 0.f};
                acc = wmma_tile64(sXT, sBt, tm, tn, lane, acc);
                const int n = tn * 16 + r;
#pragma unroll
                for (int q = 0; q < 8; ++q) {
                    stReg[u][q] = atot * stReg[u][q] + acc[q];
                    const int pr = tm * 16 + half * 8 + q;
                    sS[pr * 64 + n] = (__bf16)stReg[u][q];
                }
            }
        }
        __syncthreads();
    }
}

// ---------------- y * silu(z), RMSNorm over 1024, write bf16 ----------------
__global__ __launch_bounds__(256) void gate_rmsnorm_kernel(
    const __bf16* __restrict__ Y, const __bf16* __restrict__ ZX,
    const float* __restrict__ norm_w, __bf16* __restrict__ G, int s)
{
    const int row = blockIdx.x;
    const int b = row / LSEQ;
    const int layer = 2 * b + s;
    const long long ybase = (long long)row * DINNER;
    const long long zbase = (long long)row * DIP;

    float g[4];
    float ss = 0.f;
#pragma unroll
    for (int j = 0; j < 4; ++j) {
        const int d = threadIdx.x + j * 256;
        const float y = (float)Y[ybase + d];
        const float z = (float)ZX[zbase + d];
        const float gg = y * (z / (1.f + __expf(-z)));
        g[j] = gg;
        ss += gg * gg;
    }
    __shared__ float red[256];
    red[threadIdx.x] = ss;
    __syncthreads();
    for (int k = 128; k > 0; k >>= 1) {
        if (threadIdx.x < k) red[threadIdx.x] += red[threadIdx.x + k];
        __syncthreads();
    }
    const float scale = rsqrtf(red[0] / (float)DINNER + 1e-5f);
#pragma unroll
    for (int j = 0; j < 4; ++j) {
        const int d = threadIdx.x + j * 256;
        G[ybase + d] = (__bf16)(g[j] * scale * norm_w[(long long)layer * DINNER + d]);
    }
}

// ---------------- LayerNorm over 512, in-place f32 + bf16 copy ----------------
__global__ __launch_bounds__(256) void layernorm_kernel(
    float* __restrict__ H, const float* __restrict__ ln_w, const float* __restrict__ ln_b,
    __bf16* __restrict__ Hb)
{
    const long long base = (long long)blockIdx.x * DMODEL;
    float v[2];
    float sum = 0.f;
#pragma unroll
    for (int j = 0; j < 2; ++j) { v[j] = H[base + threadIdx.x + j * 256]; sum += v[j]; }
    __shared__ float red[256];
    red[threadIdx.x] = sum;
    __syncthreads();
    for (int k = 128; k > 0; k >>= 1) { if (threadIdx.x < k) red[threadIdx.x] += red[threadIdx.x + k]; __syncthreads(); }
    const float mu = red[0] / (float)DMODEL;
    __syncthreads();
    float vs = 0.f;
#pragma unroll
    for (int j = 0; j < 2; ++j) { const float d = v[j] - mu; vs += d * d; }
    red[threadIdx.x] = vs;
    __syncthreads();
    for (int k = 128; k > 0; k >>= 1) { if (threadIdx.x < k) red[threadIdx.x] += red[threadIdx.x + k]; __syncthreads(); }
    const float scale = rsqrtf(red[0] / (float)DMODEL + 1e-5f);
#pragma unroll
    for (int j = 0; j < 2; ++j) {
        const int d = threadIdx.x + j * 256;
        const float o = (v[j] - mu) * scale * ln_w[d] + ln_b[d];
        H[base + d] = o;
        Hb[base + d] = (__bf16)o;
    }
}

// ---------------- attention score: dot(hidden[t], w2) + b2 ----------------
__global__ __launch_bounds__(128) void attn_score_kernel(
    const float* __restrict__ Hid, const float* __restrict__ w2, const float* __restrict__ b2,
    float* __restrict__ S)
{
    const int t = blockIdx.x;
    __shared__ float red[128];
    red[threadIdx.x] = Hid[(long long)t * 128 + threadIdx.x] * w2[threadIdx.x];
    __syncthreads();
    for (int k = 64; k > 0; k >>= 1) { if (threadIdx.x < k) red[threadIdx.x] += red[threadIdx.x + k]; __syncthreads(); }
    if (threadIdx.x == 0) S[t] = red[0] + b2[0];
}

// ---------------- softmax over LTOT tokens (single block) ----------------
__global__ __launch_bounds__(1024) void softmax_kernel(float* __restrict__ S, int n) {
    __shared__ float red[1024];
    const int tid = threadIdx.x;
    float m = -3.4e38f;
    for (int i = tid; i < n; i += 1024) m = fmaxf(m, S[i]);
    red[tid] = m;
    __syncthreads();
    for (int k = 512; k > 0; k >>= 1) { if (tid < k) red[tid] = fmaxf(red[tid], red[tid + k]); __syncthreads(); }
    const float gm = red[0];
    __syncthreads();
    float sum = 0.f;
    for (int i = tid; i < n; i += 1024) sum += __expf(S[i] - gm);
    red[tid] = sum;
    __syncthreads();
    for (int k = 512; k > 0; k >>= 1) { if (tid < k) red[tid] += red[tid + k]; __syncthreads(); }
    const float inv = 1.f / red[0];
    for (int i = tid; i < n; i += 1024) S[i] = __expf(S[i] - gm) * inv;
}

// ---------------- attention pooling: pooled[d] = sum_t a[t]*h[t][d] ----------------
__global__ __launch_bounds__(256) void pool_kernel(
    const float* __restrict__ Hn, const float* __restrict__ Aw, float* __restrict__ P)
{
    const int d = blockIdx.x;
    float s = 0.f;
    for (int t = threadIdx.x; t < LTOT; t += 256)
        s += Aw[t] * Hn[(long long)t * DMODEL + d];
    __shared__ float red[256];
    red[threadIdx.x] = s;
    __syncthreads();
    for (int k = 128; k > 0; k >>= 1) { if (threadIdx.x < k) red[threadIdx.x] += red[threadIdx.x + k]; __syncthreads(); }
    if (threadIdx.x == 0) P[d] = red[0];
}

// ---------------- classifier head + softmax -> d_out[0..3] ----------------
__global__ __launch_bounds__(256) void head_kernel(
    const float* __restrict__ P, const float* __restrict__ cw, const float* __restrict__ cb,
    float* __restrict__ out)
{
    __shared__ float red[256];
    float a0 = 0.f, a1 = 0.f;
    for (int d = threadIdx.x; d < DMODEL; d += 256) {
        const float p = P[d];
        a0 += p * cw[d];
        a1 += p * cw[DMODEL + d];
    }
    red[threadIdx.x] = a0;
    __syncthreads();
    for (int k = 128; k > 0; k >>= 1) { if (threadIdx.x < k) red[threadIdx.x] += red[threadIdx.x + k]; __syncthreads(); }
    const float l0 = red[0] + cb[0];
    __syncthreads();
    red[threadIdx.x] = a1;
    __syncthreads();
    for (int k = 128; k > 0; k >>= 1) { if (threadIdx.x < k) red[threadIdx.x] += red[threadIdx.x + k]; __syncthreads(); }
    const float l1 = red[0] + cb[1];
    if (threadIdx.x == 0) {
        const float m = fmaxf(l0, l1);
        const float e0 = __expf(l0 - m), e1 = __expf(l1 - m);
        const float inv = 1.f / (e0 + e1);
        out[0] = l0; out[1] = l1; out[2] = e0 * inv; out[3] = e1 * inv;
    }
}

// ==================================================================
extern "C" void kernel_launch(void* const* d_in, const int* in_sizes, int n_in,
                              void* d_out, int out_size, void* d_ws, size_t ws_size,
                              hipStream_t stream) {
    (void)in_sizes; (void)n_in; (void)out_size; (void)ws_size;

    const float* x        = (const float*)d_in[0];
    const float* fc1_w    = (const float*)d_in[1];
    const float* fc1_b    = (const float*)d_in[2];
    const float* in_w     = (const float*)d_in[3];
    const float* conv_w   = (const float*)d_in[4];
    const float* conv_b   = (const float*)d_in[5];
    const float* dt_bias  = (const float*)d_in[6];
    const float* A_log    = (const float*)d_in[7];
    const float* Dh       = (const float*)d_in[8];
    const float* norm_w   = (const float*)d_in[9];
    const float* out_w    = (const float*)d_in[10];
    const float* ln_w     = (const float*)d_in[11];
    const float* ln_b     = (const float*)d_in[12];
    const float* attn_w1  = (const float*)d_in[13];
    const float* attn_b1  = (const float*)d_in[14];
    const float* attn_w2  = (const float*)d_in[15];
    const float* attn_b2  = (const float*)d_in[16];
    const float* cls_w    = (const float*)d_in[17];
    const float* cls_b    = (const float*)d_in[18];
    float* out = (float*)d_out;

    // ----- workspace layout (all offsets multiples of 256 bytes) -----
    char* ws = (char*)d_ws;
    const long long WIN_STRIDE  = (long long)DIP * DMODEL;      // 1,122,304
    const long long WOUT_STRIDE = (long long)DMODEL * DINNER;   //   524,288
    __bf16* WFC1 = (__bf16*)(ws + 0);                               // 512*1024 bf16
    __bf16* WIN  = (__bf16*)(ws + 1048576LL);                       // 6 * 2192*512 bf16
    __bf16* WOUT = (__bf16*)(ws + 14516224LL);                      // 6 * 512*1024 bf16
    __bf16* WAT1 = (__bf16*)(ws + 20807680LL);                      // 128*512 bf16
    float*  H3   = (float*)(ws + 20938752LL);                       // 3*10000*512 f32 (-> final concat)
    __bf16* HBF  = (__bf16*)(ws + 82378752LL);                      // 30000*512 bf16
    __bf16* ZXBF = (__bf16*)(ws + 113098752LL);                     // 3*10000*2192 bf16
    float*  XC   = (float*)(ws + 244618752LL);                      // 3*10000*1152 f32
    __bf16* YBF  = (__bf16*)(ws + 382858752LL);                     // 3*10000*1024 bf16
    float*  SMALL= (float*)(ws + 444298752LL);
    // aliases (lifetimes disjoint)
    __bf16* XBF = ZXBF;                    // x in bf16 (before in_proj uses ZXBF)
    float*  ATT = (float*)ZXBF;            // attention hidden 30000x128 (after stages)
    __bf16* GBF = (__bf16*)XC;             // gated/normed activations (after scan consumed XC)
    float*  SC  = SMALL;                   // 30000 scores
    float*  PO  = (float*)((char*)SMALL + 131072);  // pooled 512

    auto cvt = [&](const float* s, __bf16* d, long long n) {
        int blocks = (int)((n + 255) / 256); if (blocks > 65535) blocks = 65535;
        f2bf_kernel<<<blocks, 256, 0, stream>>>(s, d, n);
    };
    auto gemm = [&](const __bf16* A, const __bf16* W, const float* bias, const float* resid,
                    float* Cf, __bf16* Cb, int M, int N, int K, int act,
                    int tilesPerBatch, long long wBatchStride) {
        const int ntiles = (M / 16) * (N / 16);
        gemm_bf16_wmma<<<(ntiles + 7) / 8, 256, 0, stream>>>(
            A, W, bias, resid, Cf, Cb, M, N, K, act, tilesPerBatch, wBatchStride);
    };

    // 1) weights -> bf16
    cvt(fc1_w,   WFC1, (long long)DMODEL * 1024);
    cvt(in_w,    WIN,  6LL * WIN_STRIDE);
    cvt(out_w,   WOUT, 6LL * WOUT_STRIDE);
    cvt(attn_w1, WAT1, 128LL * DMODEL);

    // 2) fc1: H3[branch0] = relu(x @ fc1_w^T + b)
    cvt(x, XBF, (long long)LSEQ * 1024);
    gemm(XBF, WFC1, fc1_b, nullptr, H3, nullptr, LSEQ, DMODEL, 1024, /*relu*/1, 0, 0);

    // 3) build flipped + transposed branch inputs
    prep_branches_kernel<<<((LSEQ * DMODEL + 255) / 256 > 65535 ? 65535 : (LSEQ * DMODEL + 255) / 256),
                           256, 0, stream>>>(H3);

    // 4) two stages; stage s runs layer (2b+s) of all 3 branches batched
    for (int s = 0; s < 2; ++s) {
        cvt(H3, HBF, (long long)LTOT * DMODEL);
        // in_proj (batched weights): ZXBF = HBF @ in_w[2b+s]^T   (bf16 out)
        gemm(HBF, WIN + (long long)s * WIN_STRIDE, nullptr, nullptr,
             nullptr, ZXBF, LTOT, DIP, DMODEL, 0, LSEQ / 16, 2LL * WIN_STRIDE);
        // depthwise conv + silu
        conv_silu_kernel<<<65535, 256, 0, stream>>>(ZXBF, conv_w, conv_b, XC, s);
        // chunked SSD scan: 3 branches x 16 heads, all-WMMA
        ssd_chunk_kernel<<<NB * NHEADS, 256, 0, stream>>>(XC, ZXBF, dt_bias, A_log, Dh, YBF, s);
        // gate + RMSNorm -> GBF (bf16, aliases XC)
        gate_rmsnorm_kernel<<<LTOT, 256, 0, stream>>>(YBF, ZXBF, norm_w, GBF, s);
        // out_proj (batched weights) + residual, write back into H3
        gemm(GBF, WOUT + (long long)s * WOUT_STRIDE, nullptr, H3,
             H3, nullptr, LTOT, DMODEL, DINNER, 0, LSEQ / 16, 2LL * WOUT_STRIDE);
    }

    // 5) LayerNorm (in place) + bf16 copy
    layernorm_kernel<<<LTOT, 256, 0, stream>>>(H3, ln_w, ln_b, HBF);

    // 6) attention: hidden = tanh(h @ w1^T + b1) ; score = hidden @ w2 + b2 ; softmax
    gemm(HBF, WAT1, attn_b1, nullptr, ATT, nullptr, LTOT, 128, DMODEL, /*tanh*/2, 0, 0);
    attn_score_kernel<<<LTOT, 128, 0, stream>>>(ATT, attn_w2, attn_b2, SC);
    softmax_kernel<<<1, 1024, 0, stream>>>(SC, LTOT);

    // 7) pooled + classifier head
    pool_kernel<<<DMODEL, 256, 0, stream>>>(H3, SC, PO);
    head_kernel<<<1, 256, 0, stream>>>(PO, cls_w, cls_b, out);
}
